// IntraFrameSpatialMamba_89180700934824
// MI455X (gfx1250) — compile-verified, matrix operand order checked by reading, not run on hardware
//
#include <hip/hip_runtime.h>

// ---------------- problem constants ----------------
#define NBLK    4
#define DIMC    256
#define DSTATE  16
#define DCONV   4
#define DIN     512          // EXPAND * DIM
#define DTRANK  16
#define BATCH   16
#define LSEQ    512
#define BLROWS  (BATCH * LSEQ)   // 8192
#define NDBC    (DTRANK + 2 * DSTATE)  // 48

typedef __attribute__((ext_vector_type(16))) _Float16 v16h;
typedef __attribute__((ext_vector_type(8)))  _Float16 v8h;
typedef __attribute__((ext_vector_type(8)))  float    v8f;

// ---------------- small helpers ----------------
__device__ __forceinline__ float silu_f(float z) {
    return z / (1.0f + __expf(-z));
}
__device__ __forceinline__ float softplus_f(float v) {
    return v > 20.0f ? v : log1pf(__expf(v));
}

// ---------------- fp32 -> f16 convert (weights) ----------------
__global__ void f32_to_f16_kernel(const float* __restrict__ in,
                                  _Float16* __restrict__ out, int n) {
    int i = blockIdx.x * blockDim.x + threadIdx.x;
    if (i < n) out[i] = (_Float16)in[i];
}

// ---------------- LayerNorm (row = 256 elems) -> f16 ----------------
__global__ __launch_bounds__(DIMC)
void layernorm_f16_kernel(const float* __restrict__ x,
                          const float* __restrict__ w,
                          const float* __restrict__ b,
                          _Float16* __restrict__ out) {
    __shared__ float red[DIMC];
    const int row = blockIdx.x;
    const int t   = threadIdx.x;
    const float v = x[(size_t)row * DIMC + t];

    red[t] = v;
    __syncthreads();
    for (int s = DIMC / 2; s > 0; s >>= 1) {
        if (t < s) red[t] += red[t + s];
        __syncthreads();
    }
    const float mu = red[0] * (1.0f / DIMC);
    __syncthreads();

    const float d = v - mu;
    red[t] = d * d;
    __syncthreads();
    for (int s = DIMC / 2; s > 0; s >>= 1) {
        if (t < s) red[t] += red[t + s];
        __syncthreads();
    }
    const float var  = red[0] * (1.0f / DIMC);
    const float rstd = rsqrtf(var + 1e-5f);

    out[(size_t)row * DIMC + t] = (_Float16)(d * rstd * w[t] + b[t]);
}

// ---------------- register-blocked WMMA GEMM ----------------
// C(MxN) = A(MxK) * Wt(NxK)^T (+bias)(+resid)
// One wave computes a 16 x (16*NT) output strip: a single A fragment per
// k-step feeds NT back-to-back v_wmma_f32_16x16x32_f16 ops (A reuse x NT,
// independent WMMAs in flight).
//
// Fragment layout (CDNA5 ISA 7.12.2, 16-bit A 16x32):
//   lane = 0..31, row/col = lane&15, koff = (lane>=16)?8:0 (halves)
//   half h (0..15): K = (h<8 ? h : 8+h) + koff
// -> per lane, each fragment = two contiguous 16B chunks of one row
//    (global_load_b128 pairs). B mirrors A with lane holding column n.
// C/D: VGPR v holds M = v + ((lane>=16)?8:0), N = lane&15.
// ----------------
template <int NT>
__global__ __launch_bounds__(32)
void gemm_wmma_f16_kernel(const _Float16* __restrict__ A,
                          const _Float16* __restrict__ Wt,
                          const float* __restrict__ bias,
                          const float* __restrict__ resid,
                          float* __restrict__ C,
                          int M, int N, int K) {
    const int lane = threadIdx.x;
    const int r    = lane & 15;
    const int koff = (lane >> 4) << 3;          // 0 or 8 halves
    const int tm   = blockIdx.y;
    const int tn0  = blockIdx.x * NT;

    const _Float16* arow = A + (size_t)(tm * 16 + r) * K;
    const _Float16* brow[NT];
#pragma unroll
    for (int j = 0; j < NT; ++j)
        brow[j] = Wt + (size_t)((tn0 + j) * 16 + r) * K;

    v8f acc[NT];
#pragma unroll
    for (int j = 0; j < NT; ++j) acc[j] = (v8f){};

    for (int k0 = 0; k0 < K; k0 += 32) {
        // speculative prefetch of the next k-slab (dropped if OOB)
        __builtin_prefetch(arow + k0 + 32 + koff, 0, 1);

        v8h alo = *(const v8h*)(arow + k0 + koff);
        v8h ahi = *(const v8h*)(arow + k0 + 16 + koff);
        v16h a;
#pragma unroll
        for (int i = 0; i < 8; ++i) { a[i] = alo[i]; a[i + 8] = ahi[i]; }

#pragma unroll
        for (int j = 0; j < NT; ++j) {
            v8h blo = *(const v8h*)(brow[j] + k0 + koff);
            v8h bhi = *(const v8h*)(brow[j] + k0 + 16 + koff);
            v16h b;
#pragma unroll
            for (int i = 0; i < 8; ++i) { b[i] = blo[i]; b[i + 8] = bhi[i]; }
            acc[j] = __builtin_amdgcn_wmma_f32_16x16x32_f16(
                /*neg_a=*/false, a, /*neg_b=*/false, b,
                /*c_mod=*/(short)0, acc[j], /*reuse_a=*/false, /*reuse_b=*/false);
        }
    }

    const int mbase = tm * 16 + ((lane >> 4) << 3);
#pragma unroll
    for (int j = 0; j < NT; ++j) {
        const int   n  = (tn0 + j) * 16 + r;
        const float bv = bias ? bias[n] : 0.0f;
#pragma unroll
        for (int v = 0; v < 8; ++v) {
            const size_t m = (size_t)(mbase + v);
            float o = acc[j][v] + bv;
            if (resid) o += resid[m * (size_t)N + n];
            C[m * (size_t)N + n] = o;
        }
    }
}

// ---------------- depthwise causal conv (DCONV=4) + bias + SiLU ----------------
__global__ void conv_silu_kernel(const float* __restrict__ xz,
                                 const float* __restrict__ cw,
                                 const float* __restrict__ cb,
                                 float* __restrict__ xc,
                                 _Float16* __restrict__ xc16) {
    const int idx = blockIdx.x * blockDim.x + threadIdx.x;   // over BLROWS*DIN
    if (idx >= BLROWS * DIN) return;
    const int c  = idx & (DIN - 1);
    const int bl = idx >> 9;                 // DIN == 512
    const int l  = bl & (LSEQ - 1);

    float acc = cb[c];
#pragma unroll
    for (int j = 0; j < DCONV; ++j) {
        const int lj = l - (DCONV - 1) + j;
        if (lj >= 0)
            acc += cw[c * DCONV + j] *
                   xz[(size_t)(bl - (DCONV - 1) + j) * (2 * DIN) + c];
    }
    const float s = silu_f(acc);
    xc[idx]   = s;
    xc16[idx] = (_Float16)s;
}

// ---------------- dt = softplus(dbc[:, :DTRANK] @ dtw^T + dtb) ----------------
__global__ void dtproj_kernel(const float* __restrict__ dbc,
                              const float* __restrict__ dtw,
                              const float* __restrict__ dtb,
                              float* __restrict__ dt) {
    const int idx = blockIdx.x * blockDim.x + threadIdx.x;   // over BLROWS*DIN
    if (idx >= BLROWS * DIN) return;
    const int d  = idx & (DIN - 1);
    const int bl = idx >> 9;

    const float* dr = dbc + (size_t)bl * NDBC;
    const float* wr = dtw + (size_t)d * DTRANK;
    float acc = dtb[d];
#pragma unroll
    for (int r = 0; r < DTRANK; ++r) acc += dr[r] * wr[r];
    dt[idx] = softplus_f(acc);
}

// ---------------- selective scan, one thread per (b,d,n) state ----------------
// 131072 threads (4096 waves) -> 16x the latency hiding of a per-(b,d)
// mapping. The y = sum_n h*C reduction is a 4-step shuffle butterfly inside
// each 16-lane group: no LDS, no barriers in the L-loop. Fused with +u*D,
// *silu(z) and the f16 downconvert for the out-GEMM.
__global__ __launch_bounds__(256)
void scan_kernel(const float* __restrict__ dbc,
                 const float* __restrict__ dt,
                 const float* __restrict__ xc,
                 const float* __restrict__ xz,
                 const float* __restrict__ A_log,
                 const float* __restrict__ Dp,
                 _Float16* __restrict__ g16) {
    const int gt = blockIdx.x * blockDim.x + threadIdx.x;  // < BATCH*DIN*DSTATE
    const int n  = gt & (DSTATE - 1);
    const int d  = (gt >> 4) & (DIN - 1);
    const int b  = gt >> 13;                               // 4 + 9 bits

    const float An = -__expf(A_log[(size_t)d * DSTATE + n]);
    const float Dv = Dp[d];
    float h = 0.0f;

    for (int t = 0; t < LSEQ; ++t) {
        const int row = b * LSEQ + t;
        const float Bn  = dbc[(size_t)row * NDBC + DTRANK + n];
        const float Cn  = dbc[(size_t)row * NDBC + DTRANK + DSTATE + n];
        const float dtv = dt[(size_t)row * DIN + d];       // uniform per 16-lane group
        const float u   = xc[(size_t)row * DIN + d];       // uniform per 16-lane group

        h = __expf(dtv * An) * h + dtv * Bn * u;

        float p = h * Cn;                                  // butterfly sum over n
        p += __shfl_xor(p, 1, 16);
        p += __shfl_xor(p, 2, 16);
        p += __shfl_xor(p, 4, 16);
        p += __shfl_xor(p, 8, 16);

        if (n == 0) {
            const float z = xz[(size_t)row * (2 * DIN) + DIN + d];
            g16[(size_t)row * DIN + d] = (_Float16)((p + u * Dv) * silu_f(z));
        }
    }
}

// ---------------- host orchestration ----------------
extern "C" void kernel_launch(void* const* d_in, const int* in_sizes, int n_in,
                              void* d_out, int out_size, void* d_ws, size_t ws_size,
                              hipStream_t stream) {
    const float* x0    = (const float*)d_in[0];
    const float* ln_w  = (const float*)d_in[1];
    const float* ln_b  = (const float*)d_in[2];
    const float* in_w  = (const float*)d_in[3];
    const float* in_b  = (const float*)d_in[4];
    const float* cw    = (const float*)d_in[5];
    const float* cb    = (const float*)d_in[6];
    const float* xp_w  = (const float*)d_in[7];
    const float* dt_w  = (const float*)d_in[8];
    const float* dt_b  = (const float*)d_in[9];
    const float* A_log = (const float*)d_in[10];
    const float* Dp    = (const float*)d_in[11];
    const float* out_w = (const float*)d_in[12];
    float* out = (float*)d_out;

    // ---- workspace carve-up (all chunks 256B aligned) ----
    char* ws = (char*)d_ws;
    size_t off = 0;
    _Float16* w16_in  = (_Float16*)(ws + off); off += (size_t)NBLK * 2 * DIN * DIMC * 2;
    _Float16* w16_xp  = (_Float16*)(ws + off); off += (size_t)NBLK * NDBC * DIN * 2;
    _Float16* w16_out = (_Float16*)(ws + off); off += (size_t)NBLK * DIMC * DIN * 2;
    _Float16* h16     = (_Float16*)(ws + off); off += (size_t)BLROWS * DIMC * 2;
    float*    xz      = (float*)(ws + off);    off += (size_t)BLROWS * 2 * DIN * 4;
    float*    xc      = (float*)(ws + off);    off += (size_t)BLROWS * DIN * 4;
    _Float16* xc16    = (_Float16*)(ws + off); off += (size_t)BLROWS * DIN * 2;
    float*    dbc     = (float*)(ws + off);    off += (size_t)BLROWS * NDBC * 4;
    float*    dtb_buf = (float*)(ws + off);    off += (size_t)BLROWS * DIN * 4;
    _Float16* g16     = (_Float16*)(ws + off); off += (size_t)BLROWS * DIN * 2;
    (void)ws_size; (void)in_sizes; (void)n_in; (void)out_size;

    // ---- per-launch weight f16 conversion ----
    {
        int n1 = NBLK * 2 * DIN * DIMC;
        f32_to_f16_kernel<<<(n1 + 255) / 256, 256, 0, stream>>>(in_w, w16_in, n1);
        int n2 = NBLK * NDBC * DIN;
        f32_to_f16_kernel<<<(n2 + 255) / 256, 256, 0, stream>>>(xp_w, w16_xp, n2);
        int n3 = NBLK * DIMC * DIN;
        f32_to_f16_kernel<<<(n3 + 255) / 256, 256, 0, stream>>>(out_w, w16_out, n3);
    }

    for (int blk = 0; blk < NBLK; ++blk) {
        const float* xin = (blk == 0) ? x0 : out;   // residual stream, in-place across blocks

        // 1) LayerNorm -> f16
        layernorm_f16_kernel<<<BLROWS, DIMC, 0, stream>>>(
            xin, ln_w + blk * DIMC, ln_b + blk * DIMC, h16);

        // 2) in_proj: xz(8192x1024) = h16 @ in_w^T + in_b   [WMMA, 16x64/wave]
        {
            dim3 g(2 * DIN / 64, BLROWS / 16);
            gemm_wmma_f16_kernel<4><<<g, 32, 0, stream>>>(
                h16, w16_in + (size_t)blk * 2 * DIN * DIMC,
                in_b + (size_t)blk * 2 * DIN, nullptr, xz,
                BLROWS, 2 * DIN, DIMC);
        }

        // 3) depthwise causal conv + SiLU
        conv_silu_kernel<<<(BLROWS * DIN) / 256, 256, 0, stream>>>(
            xz, cw + (size_t)blk * DIN * DCONV, cb + (size_t)blk * DIN, xc, xc16);

        // 4) x_proj: dbc(8192x48) = xc16 @ xp_w^T   [WMMA, 16x48/wave]
        {
            dim3 g(NDBC / 48, BLROWS / 16);
            gemm_wmma_f16_kernel<3><<<g, 32, 0, stream>>>(
                xc16, w16_xp + (size_t)blk * NDBC * DIN,
                nullptr, nullptr, dbc, BLROWS, NDBC, DIN);
        }

        // 5) dt projection + softplus (K=16, VALU)
        dtproj_kernel<<<(BLROWS * DIN) / 256, 256, 0, stream>>>(
            dbc, dt_w + (size_t)blk * DIN * DTRANK, dt_b + (size_t)blk * DIN, dtb_buf);

        // 6) selective scan (thread per (b,d,n), shuffle reduction) -> g16
        scan_kernel<<<(BATCH * DIN * DSTATE) / 256, 256, 0, stream>>>(
            dbc, dtb_buf, xc, xz,
            A_log + (size_t)blk * DIN * DSTATE, Dp + (size_t)blk * DIN, g16);

        // 7) out_proj + residual: out = xin + g16 @ ow^T   [WMMA, 16x64/wave]
        {
            dim3 g(DIMC / 64, BLROWS / 16);
            gemm_wmma_f16_kernel<4><<<g, 32, 0, stream>>>(
                g16, w16_out + (size_t)blk * DIMC * DIN,
                nullptr, xin, out, BLROWS, DIMC, DIN);
        }
    }
}
